// TransformerLayer_47717086659013
// MI455X (gfx1250) — compile-verified, matrix operand order checked by reading, not run on hardware
//
#include <hip/hip_runtime.h>
#include <hip/hip_bf16.h>

// ---------------- dims (fixed by the reference) ----------------
#define LNUM 4
#define BB   2
#define SS   1024
#define DD   1024
#define FF   4096
#define HH   16
#define DH   64
#define MM   (BB*SS)      // 2048 token rows

typedef __attribute__((ext_vector_type(16))) __bf16 v16bf;
typedef __attribute__((ext_vector_type(8)))  __bf16 v8bf;
typedef __attribute__((ext_vector_type(8)))  float  v8f;
typedef unsigned int u32;

union BF16x16 { v16bf v; v8bf h[2]; };

static __device__ inline __bf16 f2bf(float f) {
    union { float f; unsigned u; } x; x.f = f;
    unsigned r = x.u + 0x7FFFu + ((x.u >> 16) & 1u);   // round-to-nearest-even
    unsigned short hs = (unsigned short)(r >> 16);
    __bf16 out; __builtin_memcpy(&out, &hs, 2);
    return out;
}

static __device__ inline v8f wmma_bf16(v16bf a, v16bf b, v8f c) {
    return __builtin_amdgcn_wmma_f32_16x16x32_bf16(
        false, a, false, b, (short)0, c, false, false);
}

// CDNA5 async copy: 16B global -> LDS per lane (ASYNCcnt tracked).
// LDS operand = wave-relative byte address (low 32 bits of generic LDS ptr).
static __device__ inline void async_g2l_b128(u32 ldsOff, const void* gptr) {
    asm volatile("global_load_async_to_lds_b128 %0, %1, off"
                 :: "v"(ldsOff), "v"(gptr) : "memory");
}
#define WAIT_ASYNC(n) asm volatile("s_wait_asynccnt " #n ::: "memory")
#define WAIT_DS0()    asm volatile("s_wait_dscnt 0" ::: "memory")

static __device__ inline u32 lds_off(const void* p) {
    return (u32)(unsigned long long)p;
}

// ---------------- per-layer fp32 -> bf16 weight conversion ----------------
__global__ __launch_bounds__(256) void wconv_kernel(
    const float* __restrict__ src, __bf16* __restrict__ dst, int n8)
{
    int i = blockIdx.x * 256 + threadIdx.x;
    if (i < n8) {
        const float4* s = (const float4*)src;
        float4 a = s[2*i], b = s[2*i + 1];
        union { v8bf v; __bf16 e[8]; } u;
        u.e[0] = f2bf(a.x); u.e[1] = f2bf(a.y); u.e[2] = f2bf(a.z); u.e[3] = f2bf(a.w);
        u.e[4] = f2bf(b.x); u.e[5] = f2bf(b.y); u.e[6] = f2bf(b.z); u.e[7] = f2bf(b.w);
        *(v8bf*)&dst[8*i] = u.v;
    }
}

// ---------------- LayerNorm: fp32 row -> bf16 row ----------------
__global__ __launch_bounds__(256) void ln_kernel(
    const float* __restrict__ x, const float* __restrict__ w,
    const float* __restrict__ b, __bf16* __restrict__ out)
{
    const int row = blockIdx.x;
    const int t   = threadIdx.x;
    const float* xr = x + (size_t)row * DD;
    float vals[4];
    float s = 0.f;
#pragma unroll
    for (int i = 0; i < 4; ++i) { vals[i] = xr[t + 256*i]; s += vals[i]; }

    __shared__ float red[256];
    red[t] = s; __syncthreads();
    for (int off = 128; off > 0; off >>= 1) {
        if (t < off) red[t] += red[t + off];
        __syncthreads();
    }
    const float mean = red[0] * (1.f / DD);
    __syncthreads();
    float s2 = 0.f;
#pragma unroll
    for (int i = 0; i < 4; ++i) { float d = vals[i] - mean; s2 += d * d; }
    red[t] = s2; __syncthreads();
    for (int off = 128; off > 0; off >>= 1) {
        if (t < off) red[t] += red[t + off];
        __syncthreads();
    }
    const float rstd = rsqrtf(red[0] * (1.f / DD) + 1e-5f);
#pragma unroll
    for (int i = 0; i < 4; ++i) {
        int idx = t + 256*i;
        out[(size_t)row * DD + idx] = f2bf((vals[i] - mean) * rstd * w[idx] + b[idx]);
    }
}

// ---------------- Tiled bf16 GEMM, double-buffered, async A tile ----------
// C[M,N] = act( A_bf16[M,K] @ B_bf16[K,N] + bias ) + resid
#define BM 128
#define BN 128
#define BK 32
#define ASTR 40   // LDS row stride (80B, 16B-aligned, conflict pad)

template<bool HASBIAS, bool GELU, bool RESID, bool OUTF>
__global__ __launch_bounds__(256) void gemm_bf16(
    const __bf16* __restrict__ A, const __bf16* __restrict__ Bw,
    const float* __restrict__ bias, const float* __restrict__ resid,
    float* __restrict__ outF, __bf16* __restrict__ outB,
    int Ndim, int Kdim)
{
    __shared__ __align__(16) __bf16 As[2][BM * ASTR];
    __shared__ __align__(16) __bf16 Bs[2][BN * ASTR];

    const int t    = threadIdx.x;
    const int lane = t & 31;
    const int wid  = t >> 5;
    const int wm   = (wid & 3) * 32;
    const int wn   = (wid >> 2) * 64;
    const int row0 = blockIdx.y * BM;
    const int col0 = blockIdx.x * BN;
    const int lr   = lane & 15;
    const int k0   = (lane >> 4) * 8;

    // per-thread tile-copy coordinates
    const int ar[2]  = { (t + 0)   >> 2, (t + 256) >> 2 };      // A rows
    const int akc[2] = { ((t + 0) & 3) * 8, ((t + 256) & 3) * 8 };
    const int bk[2]  = { (t + 0)   >> 4, (t + 256) >> 4 };      // B k
    const int bn8[2] = { ((t + 0) & 15) * 8, ((t + 256) & 15) * 8 };

    v8f acc[2][4] = {};
    v8bf bregs[2];
    const int nsteps = Kdim / BK;

    // prologue: tile 0 -> As[0] (async) + B regs
#pragma unroll
    for (int i = 0; i < 2; ++i)
        async_g2l_b128(lds_off(&As[0][ar[i]*ASTR + akc[i]]),
                       &A[(size_t)(row0 + ar[i]) * Kdim + akc[i]]);
#pragma unroll
    for (int i = 0; i < 2; ++i)
        bregs[i] = *(const v8bf*)&Bw[(size_t)bk[i] * Ndim + col0 + bn8[i]];

    for (int s = 0; s < nsteps; ++s) {
        const int cur = s & 1, nxt = cur ^ 1;
        // B tile s: regs -> LDS (transposed to N-major)
#pragma unroll
        for (int i = 0; i < 2; ++i)
#pragma unroll
            for (int j = 0; j < 8; ++j)
                Bs[cur][(bn8[i] + j) * ASTR + bk[i]] = bregs[i][j];
        // prefetch B tile s+1 into regs (overlaps barrier + compute)
        if (s + 1 < nsteps) {
            const int kk = (s + 1) * BK;
#pragma unroll
            for (int i = 0; i < 2; ++i)
                bregs[i] = *(const v8bf*)&Bw[(size_t)(kk + bk[i]) * Ndim + col0 + bn8[i]];
        }
        WAIT_ASYNC(0);          // A tile s resident in LDS
        __syncthreads();
        // issue A tile s+1 async AFTER barrier (no readers of As[nxt] remain)
        if (s + 1 < nsteps) {
            const int kk = (s + 1) * BK;
#pragma unroll
            for (int i = 0; i < 2; ++i)
                async_g2l_b128(lds_off(&As[nxt][ar[i]*ASTR + akc[i]]),
                               &A[(size_t)(row0 + ar[i]) * Kdim + kk + akc[i]]);
        }
        // ---- compute on buffers[cur] ----
        BF16x16 afr[2], bfr[4];
#pragma unroll
        for (int mt = 0; mt < 2; ++mt) {
            int r = wm + mt*16 + lr;
            afr[mt].h[0] = *(const v8bf*)&As[cur][r*ASTR + k0];
            afr[mt].h[1] = *(const v8bf*)&As[cur][r*ASTR + k0 + 16];
        }
#pragma unroll
        for (int nt = 0; nt < 4; ++nt) {
            int n = wn + nt*16 + lr;
            bfr[nt].h[0] = *(const v8bf*)&Bs[cur][n*ASTR + k0];
            bfr[nt].h[1] = *(const v8bf*)&Bs[cur][n*ASTR + k0 + 16];
        }
#pragma unroll
        for (int mt = 0; mt < 2; ++mt)
#pragma unroll
            for (int nt = 0; nt < 4; ++nt)
                acc[mt][nt] = wmma_bf16(afr[mt].v, bfr[nt].v, acc[mt][nt]);
    }

    // ---- epilogue (compile-time specialized, branch-free) ----
    const int rsel = (lane >> 4) * 8;
#pragma unroll
    for (int mt = 0; mt < 2; ++mt)
#pragma unroll
        for (int nt = 0; nt < 4; ++nt) {
            const int gn = col0 + wn + nt*16 + lr;
            const float bv = HASBIAS ? bias[gn] : 0.f;
#pragma unroll
            for (int v = 0; v < 8; ++v) {
                const int gm = row0 + wm + mt*16 + rsel + v;
                float val = acc[mt][nt][v] + bv;
                if (GELU)  val = 0.5f * val * (1.0f + erff(val * 0.70710678118f));
                if (RESID) val += resid[(size_t)gm * Ndim + gn];
                if (OUTF)  outF[(size_t)gm * Ndim + gn] = val;
                else       outB[(size_t)gm * Ndim + gn] = f2bf(val);
            }
        }
}

// ---------------- Flash attention (bf16 in, bf16 out) ----------------
#define QT   128
#define CT   64
#define QSTR 72   // 144B row stride, 16B aligned

__global__ __launch_bounds__(256) void attn_kernel(
    const __bf16* __restrict__ Q, const __bf16* __restrict__ K,
    const __bf16* __restrict__ V, __bf16* __restrict__ O)
{
    __shared__ __align__(16) __bf16 Qs[QT * QSTR];
    __shared__ __align__(16) __bf16 Ks[CT * QSTR];
    __shared__ __align__(16) __bf16 Vt[DH * QSTR];       // V transposed [dh][kv]
    __shared__ __align__(16) __bf16 Ps[8 * 16 * QSTR];   // per-wave P staging

    const int t    = threadIdx.x;
    const int lane = t & 31;
    const int wid  = t >> 5;
    const int bq   = blockIdx.z;
    const int hh   = blockIdx.y;
    const int qt   = blockIdx.x;
    const size_t base = (size_t)bq * SS * DD + (size_t)hh * DH;
    const int lr = lane & 15;
    const int k0 = (lane >> 4) * 8;

    // ---- Q tile 128x64 (async to LDS) ----
#pragma unroll
    for (int i = 0; i < 4; ++i) {
        int c = t + 256*i;
        int r = c >> 3;
        int c8 = (c & 7) * 8;
        async_g2l_b128(lds_off(&Qs[r*QSTR + c8]),
                       &Q[base + (size_t)(qt*QT + r) * DD + c8]);
    }
    WAIT_ASYNC(0);
    __syncthreads();

    BF16x16 qf[2];
    {
        int r = wid*16 + lr;
        qf[0].h[0] = *(const v8bf*)&Qs[r*QSTR + k0];
        qf[0].h[1] = *(const v8bf*)&Qs[r*QSTR + k0 + 16];
        qf[1].h[0] = *(const v8bf*)&Qs[r*QSTR + 32 + k0];
        qf[1].h[1] = *(const v8bf*)&Qs[r*QSTR + 32 + k0 + 16];
    }

    v8f oacc[4] = {};
    float mrun[8], lrun[8];
#pragma unroll
    for (int v = 0; v < 8; ++v) { mrun[v] = -3.0e38f; lrun[v] = 0.f; }
    const float scale = 0.125f;   // 1/sqrt(64)
    const int pb = wid * 16 * QSTR;

    for (int c0 = 0; c0 < SS; c0 += CT) {
        __syncthreads();
        // K chunk 64x64 row-major (async)
#pragma unroll
        for (int i = 0; i < 2; ++i) {
            int c = t + 256*i;
            int r = c >> 3;
            int c8 = (c & 7) * 8;
            async_g2l_b128(lds_off(&Ks[r*QSTR + c8]),
                           &K[base + (size_t)(c0 + r) * DD + c8]);
        }
        // V chunk 64x64 transposed [dh][kv] (scalar stores)
#pragma unroll
        for (int i = 0; i < 16; ++i) {
            int c  = t + 256*i;
            int kv = c >> 6;
            int dh = c & 63;
            Vt[dh*QSTR + kv] = V[base + (size_t)(c0 + kv) * DD + dh];
        }
        WAIT_ASYNC(0);
        __syncthreads();

        // ---- S = Q K^T ----
        v8f sc[4] = {};
#pragma unroll
        for (int nt = 0; nt < 4; ++nt) {
            BF16x16 bf_;
            int n = nt*16 + lr;
            bf_.h[0] = *(const v8bf*)&Ks[n*QSTR + k0];
            bf_.h[1] = *(const v8bf*)&Ks[n*QSTR + k0 + 16];
            sc[nt] = wmma_bf16(qf[0].v, bf_.v, sc[nt]);
            bf_.h[0] = *(const v8bf*)&Ks[n*QSTR + 32 + k0];
            bf_.h[1] = *(const v8bf*)&Ks[n*QSTR + 32 + k0 + 16];
            sc[nt] = wmma_bf16(qf[1].v, bf_.v, sc[nt]);
        }

        // ---- online softmax (rows live in 16-lane halves) ----
#pragma unroll
        for (int v = 0; v < 8; ++v) {
            float s0 = sc[0][v]*scale, s1 = sc[1][v]*scale;
            float s2 = sc[2][v]*scale, s3 = sc[3][v]*scale;
            float mx = fmaxf(fmaxf(s0, s1), fmaxf(s2, s3));
#pragma unroll
            for (int off = 1; off < 16; off <<= 1)
                mx = fmaxf(mx, __shfl_xor(mx, off, 32));
            float mn = fmaxf(mrun[v], mx);
            float al = __expf(mrun[v] - mn);
            float p0 = __expf(s0 - mn), p1 = __expf(s1 - mn);
            float p2 = __expf(s2 - mn), p3 = __expf(s3 - mn);
            float rs = p0 + p1 + p2 + p3;
#pragma unroll
            for (int off = 1; off < 16; off <<= 1)
                rs += __shfl_xor(rs, off, 32);
            lrun[v] = lrun[v] * al + rs;
            mrun[v] = mn;
            oacc[0][v] *= al; oacc[1][v] *= al;
            oacc[2][v] *= al; oacc[3][v] *= al;
            int m = (lane >> 4) * 8 + v;
            __bf16* pr = &Ps[pb + m*QSTR + lr];
            pr[0]  = f2bf(p0); pr[16] = f2bf(p1);
            pr[32] = f2bf(p2); pr[48] = f2bf(p3);
        }
        WAIT_DS0();   // wave-private staging; LDS in-order per wave + fence

        // ---- O += P V ----
        BF16x16 pf[2];
        {
            int m = lr;
            pf[0].h[0] = *(const v8bf*)&Ps[pb + m*QSTR + k0];
            pf[0].h[1] = *(const v8bf*)&Ps[pb + m*QSTR + k0 + 16];
            pf[1].h[0] = *(const v8bf*)&Ps[pb + m*QSTR + 32 + k0];
            pf[1].h[1] = *(const v8bf*)&Ps[pb + m*QSTR + 32 + k0 + 16];
        }
#pragma unroll
        for (int ot = 0; ot < 4; ++ot) {
            BF16x16 bf_;
            int n = ot*16 + lr;
            bf_.h[0] = *(const v8bf*)&Vt[n*QSTR + k0];
            bf_.h[1] = *(const v8bf*)&Vt[n*QSTR + k0 + 16];
            oacc[ot] = wmma_bf16(pf[0].v, bf_.v, oacc[ot]);
            bf_.h[0] = *(const v8bf*)&Vt[n*QSTR + 32 + k0];
            bf_.h[1] = *(const v8bf*)&Vt[n*QSTR + 32 + k0 + 16];
            oacc[ot] = wmma_bf16(pf[1].v, bf_.v, oacc[ot]);
        }
    }

    // ---- write O (normalize) ----
#pragma unroll
    for (int v = 0; v < 8; ++v) {
        float inv = 1.f / lrun[v];
        int m = (lane >> 4) * 8 + v;
        size_t rowg = (size_t)(bq*SS + qt*QT + wid*16 + m) * DD + hh*DH;
#pragma unroll
        for (int ot = 0; ot < 4; ++ot)
            O[rowg + ot*16 + lr] = f2bf(oacc[ot][v] * inv);
    }
}

// ---------------- host-side orchestration ----------------
extern "C" void kernel_launch(void* const* d_in, const int* in_sizes, int n_in,
                              void* d_out, int out_size, void* d_ws, size_t ws_size,
                              hipStream_t stream) {
    const float* x    = (const float*)d_in[0];
    const float* ln1w = (const float*)d_in[1];
    const float* ln1b = (const float*)d_in[2];
    const float* Wq   = (const float*)d_in[3];
    const float* Wk   = (const float*)d_in[4];
    const float* Wv   = (const float*)d_in[5];
    const float* Wo   = (const float*)d_in[6];
    const float* bo   = (const float*)d_in[7];
    const float* ln2w = (const float*)d_in[8];
    const float* ln2b = (const float*)d_in[9];
    const float* W1   = (const float*)d_in[10];
    const float* b1   = (const float*)d_in[11];
    const float* W2   = (const float*)d_in[12];
    const float* b2   = (const float*)d_in[13];

    float* xo = (float*)d_out;                      // running activation x
    const size_t MD = (size_t)MM * DD;
    const size_t DD2 = (size_t)DD * DD;
    const size_t DF  = (size_t)DD * FF;
    __bf16* h  = (__bf16*)d_ws;                     // LN output   (MD)
    __bf16* qb = h  + MD;                           // Q bf16      (MD)
    __bf16* kb = qb + MD;                           // K bf16      (MD)
    __bf16* vb = kb + MD;                           // V bf16      (MD)
    __bf16* ob = vb + MD;                           // attn out    (MD)
    __bf16* ub = ob + MD;                           // FFN hidden  (MM*FF)
    __bf16* wb = ub + (size_t)MM * FF;              // per-layer bf16 weights (12M)
    __bf16* wq_b = wb;
    __bf16* wk_b = wq_b + DD2;
    __bf16* wv_b = wk_b + DD2;
    __bf16* wo_b = wv_b + DD2;
    __bf16* w1_b = wo_b + DD2;
    __bf16* w2_b = w1_b + DF;

    hipMemcpyAsync(xo, x, sizeof(float) * MD, hipMemcpyDeviceToDevice, stream);

    const dim3 blk(256);
    const dim3 gD(DD / BN, MM / BM);    // (8, 16)
    const dim3 gF(FF / BN, MM / BM);    // (32, 16)
    const dim3 gA(SS / QT, HH, BB);     // (8, 16, 2)
    const int cD = (int)(DD2 / 8 / 256);   // wconv blocks for DxD
    const int cF = (int)(DF  / 8 / 256);   // wconv blocks for DxF

    for (int l = 0; l < LNUM; ++l) {
        // convert this layer's weights fp32 -> bf16 (hits L2, negligible time)
        wconv_kernel<<<cD, blk, 0, stream>>>(Wq + (size_t)l*DD2, wq_b, (int)(DD2/8));
        wconv_kernel<<<cD, blk, 0, stream>>>(Wk + (size_t)l*DD2, wk_b, (int)(DD2/8));
        wconv_kernel<<<cD, blk, 0, stream>>>(Wv + (size_t)l*DD2, wv_b, (int)(DD2/8));
        wconv_kernel<<<cD, blk, 0, stream>>>(Wo + (size_t)l*DD2, wo_b, (int)(DD2/8));
        wconv_kernel<<<cF, blk, 0, stream>>>(W1 + (size_t)l*DF,  w1_b, (int)(DF/8));
        wconv_kernel<<<cF, blk, 0, stream>>>(W2 + (size_t)l*DF,  w2_b, (int)(DF/8));

        // --- attention sub-block ---
        ln_kernel<<<MM, blk, 0, stream>>>(xo, ln1w + l*DD, ln1b + l*DD, h);
        gemm_bf16<false,false,false,false><<<gD, blk, 0, stream>>>(
            h, wq_b, nullptr, nullptr, nullptr, qb, DD, DD);
        gemm_bf16<false,false,false,false><<<gD, blk, 0, stream>>>(
            h, wk_b, nullptr, nullptr, nullptr, kb, DD, DD);
        gemm_bf16<false,false,false,false><<<gD, blk, 0, stream>>>(
            h, wv_b, nullptr, nullptr, nullptr, vb, DD, DD);
        attn_kernel<<<gA, blk, 0, stream>>>(qb, kb, vb, ob);
        gemm_bf16<true,false,true,true><<<gD, blk, 0, stream>>>(
            ob, wo_b, bo + l*DD, xo, xo, nullptr, DD, DD);   // x += o@Wo + bo
        // --- FFN sub-block ---
        ln_kernel<<<MM, blk, 0, stream>>>(xo, ln2w + l*DD, ln2b + l*DD, h);
        gemm_bf16<true,true,false,false><<<gF, blk, 0, stream>>>(
            h, w1_b, b1 + (size_t)l*FF, nullptr, nullptr, ub, FF, DD);  // GELU
        gemm_bf16<true,false,true,true><<<gD, blk, 0, stream>>>(
            ub, w2_b, b2 + l*DD, xo, xo, nullptr, DD, FF);   // x += gelu@W2 + b2
    }
}
